// RGCNLinkPredictor_89842125898104
// MI455X (gfx1250) — matrix operand order repaired and checked by reading, not compiled, not run on hardware
//
#include <hip/hip_runtime.h>
#include <math.h>

// Problem constants (match reference)
#define N_ENT   50000
#define N_REL   8
#define EMB     256
#define N_BASES 4
#define HALF_D  128
#define K_MEAN  (N_REL * EMB)   // 2048
#define K_TOT   (K_MEAN + EMB)  // 2304
#define TM      128             // GEMM block row tile

typedef __attribute__((ext_vector_type(16))) __bf16 v16bf;
typedef __attribute__((ext_vector_type(8)))  float  v8f;

// ---------------------------------------------------------------------------
// 1) Edge scatter: agg[(dst*R+etype)*256 + c] += x[src*256 + c]; cnt[key] += 1
//    One 256-thread block per edge; x rows are L2-resident (51 MB < 192 MB L2).
// ---------------------------------------------------------------------------
__global__ __launch_bounds__(EMB) void rgcn_scatter(
    const int* __restrict__ ei,   // (2, E) flattened: [0..E)=src, [E..2E)=dst
    const int* __restrict__ et,   // (E,)
    const float* __restrict__ x,  // (N, 256)
    float* __restrict__ agg,      // (N*R, 256)
    float* __restrict__ cnt,      // (N*R,)
    int E)
{
  int e = blockIdx.x;
  if (e >= E) return;
  int src = ei[e];
  int dst = ei[E + e];
  int r   = et[e];
  int key = dst * N_REL + r;
  float v = x[(size_t)src * EMB + threadIdx.x];
  __hip_atomic_fetch_add(agg + (size_t)key * EMB + threadIdx.x, v,
                         __ATOMIC_RELAXED, __HIP_MEMORY_SCOPE_AGENT);
  if (threadIdx.x == 0)
    __hip_atomic_fetch_add(cnt + key, 1.0f,
                           __ATOMIC_RELAXED, __HIP_MEMORY_SCOPE_AGENT);
}

// ---------------------------------------------------------------------------
// 2) Fused mean (divide by clamp(cnt,1)) + cast to bf16: A-matrix "mean" part.
// ---------------------------------------------------------------------------
__global__ void rgcn_meanbf(const float* __restrict__ agg,
                            const float* __restrict__ cnt,
                            __bf16* __restrict__ meanbf)
{
  const size_t total = (size_t)N_ENT * N_REL * EMB;
  for (size_t i = (size_t)blockIdx.x * blockDim.x + threadIdx.x; i < total;
       i += (size_t)gridDim.x * blockDim.x) {
    float c = cnt[i >> 8];
    float s = 1.0f / fmaxf(c, 1.0f);
    meanbf[i] = (__bf16)(agg[i] * s);
  }
}

// 2b) Cast current layer input x to bf16: A-matrix "root" part.
__global__ void rgcn_xbf(const float* __restrict__ xin,
                         __bf16* __restrict__ xbf)
{
  const size_t total = (size_t)N_ENT * EMB;
  for (size_t i = (size_t)blockIdx.x * blockDim.x + threadIdx.x; i < total;
       i += (size_t)gridDim.x * blockDim.x)
    xbf[i] = (__bf16)xin[i];
}

// ---------------------------------------------------------------------------
// 3) Build stacked weight matrix, TRANSPOSED as BwT[o][k] (256 x 2304, bf16):
//    rows k = r*256+i  -> W_r[i][o] = sum_b comp[r,b] * bases[b][i][o]
//    rows k = 2048+i   -> root[i][o]
//    Transposed layout makes WMMA B-fragment loads contiguous in k.
// ---------------------------------------------------------------------------
__global__ __launch_bounds__(EMB) void rgcn_weights(
    const float* __restrict__ comp,   // (R, NB)
    const float* __restrict__ bases,  // (NB, 256, 256)
    const float* __restrict__ root,   // (256, 256)
    __bf16* __restrict__ BwT)         // (256, 2304)  [o][k]
{
  int k = blockIdx.x;    // 0..2303
  int o = threadIdx.x;   // 0..255
  float v;
  if (k < K_MEAN) {
    int r = k >> 8, i = k & 255;
    v = 0.0f;
    for (int b = 0; b < N_BASES; ++b)
      v += comp[r * N_BASES + b] * bases[((size_t)b * EMB + i) * EMB + o];
  } else {
    v = root[(size_t)(k - K_MEAN) * EMB + o];
  }
  BwT[(size_t)o * K_TOT + k] = (__bf16)v;
}

// ---------------------------------------------------------------------------
// 4) WMMA GEMM:  out(N x 256) = [meanbf | xbf](N x 2304) * Bw(2304 x 256)
//                 + bias, optional ReLU.
//    Block = 256 threads (8 wave32), tile 128(M) x 256(N).
//    Wave (w&3, w>>2) owns a 32x128 sub-tile = 2x8 v_wmma_f32_16x16x32_bf16
//    accumulators. B (1.2 MB) stays hot in L2; no LDS staging required.
//    Fragment lane layouts per CDNA5 ISA 7.12.2:
//      A 16x32 bf16 : lane<16 row M=lane, elems0-7 k=0..7, elems8-15 k=16..23
//                     lane>=16 same row, k+8 (i.e. 8..15 / 24..31)
//      B 32x16 bf16 : lane&15 = col N, elems k order same as A (0..7,16..23; +8 for lane>=16)
//      C/D 16x16 f32: N = lane&15, M = v + (lane>=16 ? 8 : 0)
// ---------------------------------------------------------------------------
__global__ __launch_bounds__(256) void rgcn_gemm(
    const __bf16* __restrict__ Am,    // (N, 2048) mean part
    const __bf16* __restrict__ Ax,    // (N, 256)  root part
    const __bf16* __restrict__ BwT,   // (256, 2304) transposed weights
    const float*  __restrict__ bias,  // (256,)
    float* __restrict__ out,          // (N, 256)
    int Nrows, int doRelu)
{
  const int lane  = threadIdx.x & 31;
  const int wave  = threadIdx.x >> 5;
  const int mBase = blockIdx.x * TM;
  const int wm    = (wave & 3) * 32;    // wave's row offset inside tile
  const int wn    = (wave >> 2) * 128;  // wave's col offset

  const int fm  = lane & 15;
  const int hi  = lane >> 4;    // 0|1
  const int fkb = hi << 3;      // A: k base for elems 0..7 (0 or 8)
  const int fko = hi << 4;      // unused (0 or 16); B uses fkb
  const int fn  = lane & 15;
  const int cm  = hi << 3;      // C/D: row offset (0 or 8)

  v8f acc[2][8];
  {
    v8f z = {0.f, 0.f, 0.f, 0.f, 0.f, 0.f, 0.f, 0.f};
    for (int i = 0; i < 2; ++i)
      for (int j = 0; j < 8; ++j) acc[i][j] = z;
  }

  // Clamp fragment row indices so tail-block loads stay in-bounds
  // (stores below are exactly guarded, so clamped rows never escape).
  int ar0 = mBase + wm + fm;
  int ar1 = ar0 + 16;
  if (ar0 >= Nrows) ar0 = Nrows - 1;
  if (ar1 >= Nrows) ar1 = Nrows - 1;

  auto mac = [&](const __bf16* __restrict__ A, int lda, int k0, int krow) {
    v16bf a[2];
    const __bf16* p0 = A + (size_t)ar0 * lda + k0 + fkb;
    const __bf16* p1 = A + (size_t)ar1 * lda + k0 + fkb;
    ((uint4*)&a[0])[0] = *(const uint4*)(p0);        // elems 0..7  (k=fkb..+7)
    ((uint4*)&a[0])[1] = *(const uint4*)(p0 + 16);   // elems 8..15 (k=fkb+16..+23)
    ((uint4*)&a[1])[0] = *(const uint4*)(p1);
    ((uint4*)&a[1])[1] = *(const uint4*)(p1 + 16);
    __builtin_prefetch(p0 + 32, 0, 0);               // global_prefetch_b8 next k-step

    for (int j = 0; j < 8; ++j) {
      const __bf16* pb = BwT + (size_t)(wn + j * 16 + fn) * K_TOT + krow + fkb;
      v16bf b;
      ((uint4*)&b)[0] = *(const uint4*)(pb);         // k = fkb..fkb+7
      ((uint4*)&b)[1] = *(const uint4*)(pb + 16);    // k = fkb+16..fkb+23
      acc[0][j] = __builtin_amdgcn_wmma_f32_16x16x32_bf16(
          false, a[0], false, b, (short)0, acc[0][j], false, false);
      acc[1][j] = __builtin_amdgcn_wmma_f32_16x16x32_bf16(
          false, a[1], false, b, (short)0, acc[1][j], false, false);
    }
  };

  for (int k0 = 0; k0 < K_MEAN; k0 += 32) mac(Am, K_MEAN, k0, k0);           // mean part
  for (int k0 = 0; k0 < EMB;    k0 += 32) mac(Ax, EMB,    k0, K_MEAN + k0);  // root part

  for (int i = 0; i < 2; ++i) {
    int rbase = mBase + wm + i * 16 + cm;
    for (int v = 0; v < 8; ++v) {
      int row = rbase + v;
      if (row >= Nrows) continue;
      float* orow = out + (size_t)row * EMB;
      for (int j = 0; j < 8; ++j) {
        int col = wn + j * 16 + fn;
        float val = acc[i][j][v] + bias[col];
        if (doRelu) val = fmaxf(val, 0.f);
        orow[col] = val;
      }
    }
  }
}

// ---------------------------------------------------------------------------
// 5) RotatE scoring: one wave32 per triple, 4 complex dims per lane,
//    wave shuffle reduction.
// ---------------------------------------------------------------------------
__global__ __launch_bounds__(256) void rotate_score(
    const float* __restrict__ x, const float* __restrict__ relemb,
    const int* __restrict__ subj, const int* __restrict__ rel,
    const int* __restrict__ obj, float* __restrict__ out, int B)
{
  int lane = threadIdx.x & 31;
  int t = blockIdx.x * 8 + (threadIdx.x >> 5);
  if (t >= B) return;
  const float* s  = x + (size_t)subj[t] * EMB;
  const float* o  = x + (size_t)obj[t] * EMB;
  const float* ph = relemb + (size_t)rel[t] * HALF_D;
  float acc = 0.f;
  for (int d = lane; d < HALF_D; d += 32) {
    float p = ph[d];
    float sr, cr;
    sincosf(p, &sr, &cr);
    float sre = s[d], sim = s[d + HALF_D];
    float ore = o[d], oim = o[d + HALF_D];
    float dre = sre * cr - sim * sr - ore;
    float dim = sre * sr + sim * cr - oim;
    acc += dre * dre + dim * dim;
  }
  for (int off = 16; off > 0; off >>= 1)
    acc += __shfl_down(acc, off, 32);
  if (lane == 0) out[t] = -acc;
}

// ---------------------------------------------------------------------------
// Host orchestration. Workspace budget ~745 MB:
//   agg 410MB | cnt 1.6MB | meanbf 205MB | xbf 26MB | BwT 1.2MB | x0,x1 51MB ea
// ---------------------------------------------------------------------------
extern "C" void kernel_launch(void* const* d_in, const int* in_sizes, int n_in,
                              void* d_out, int out_size, void* d_ws, size_t ws_size,
                              hipStream_t stream)
{
  const int*   edge_index   = (const int*)  d_in[0];
  const int*   edge_type    = (const int*)  d_in[1];
  const int*   pos_subj     = (const int*)  d_in[2];
  const int*   pos_rel      = (const int*)  d_in[3];
  const int*   pos_obj      = (const int*)  d_in[4];
  const int*   neg_subj     = (const int*)  d_in[5];
  const int*   neg_rel      = (const int*)  d_in[6];
  const int*   neg_obj      = (const int*)  d_in[7];
  const float* entity_emb   = (const float*)d_in[8];
  const float* relation_emb = (const float*)d_in[9];
  const float* comp         = (const float*)d_in[10];
  const float* bases        = (const float*)d_in[11];
  const float* root         = (const float*)d_in[12];
  const float* bias         = (const float*)d_in[13];

  const int E  = in_sizes[1];
  const int Bp = in_sizes[2];
  const int Bn = in_sizes[5];

  char* ws = (char*)d_ws;
  size_t off = 0;
  auto carve = [&](size_t bytes) -> char* {
    char* p = ws + off;
    off += (bytes + 255) & ~(size_t)255;
    return p;
  };

  float*  agg    = (float*) carve((size_t)N_ENT * N_REL * EMB * sizeof(float));
  float*  cnt    = (float*) carve((size_t)N_ENT * N_REL * sizeof(float));
  __bf16* meanbf = (__bf16*)carve((size_t)N_ENT * K_MEAN * sizeof(__bf16));
  __bf16* xbf    = (__bf16*)carve((size_t)N_ENT * EMB * sizeof(__bf16));
  __bf16* BwT    = (__bf16*)carve((size_t)EMB * K_TOT * sizeof(__bf16));
  float*  xb0    = (float*) carve((size_t)N_ENT * EMB * sizeof(float));
  float*  xb1    = (float*) carve((size_t)N_ENT * EMB * sizeof(float));

  const float* xin = entity_emb;
  for (int l = 0; l < 2; ++l) {
    hipMemsetAsync(agg, 0, (size_t)N_ENT * N_REL * EMB * sizeof(float), stream);
    hipMemsetAsync(cnt, 0, (size_t)N_ENT * N_REL * sizeof(float), stream);

    rgcn_scatter<<<E, EMB, 0, stream>>>(edge_index, edge_type, xin, agg, cnt, E);
    rgcn_meanbf<<<8192, 256, 0, stream>>>(agg, cnt, meanbf);
    rgcn_xbf<<<4096, 256, 0, stream>>>(xin, xbf);
    rgcn_weights<<<K_TOT, EMB, 0, stream>>>(
        comp + (size_t)l * N_REL * N_BASES,
        bases + (size_t)l * N_BASES * EMB * EMB,
        root + (size_t)l * EMB * EMB, BwT);

    float* xout = (l == 0) ? xb0 : xb1;
    rgcn_gemm<<<(N_ENT + TM - 1) / TM, 256, 0, stream>>>(
        meanbf, xbf, BwT, bias + (size_t)l * EMB, xout, N_ENT, l == 0 ? 1 : 0);
    xin = xout;
  }

  float* outp = (float*)d_out;
  rotate_score<<<(Bp + 7) / 8, 256, 0, stream>>>(
      xb1, relation_emb, pos_subj, pos_rel, pos_obj, outp, Bp);
  rotate_score<<<(Bn + 7) / 8, 256, 0, stream>>>(
      xb1, relation_emb, neg_subj, neg_rel, neg_obj, outp + Bp, Bn);
}
